// TransformerBlock_14448269984095
// MI455X (gfx1250) — compile-verified, matrix operand order checked by reading, not run on hardware
//
// MI455X (gfx1250, CDNA5) transformer block: nGPT-style attention+FFN with AdaFM
// modulation. Heavy math on v_wmma_f32_16x16x32_bf16 (wave32); global->LDS tile
// staging via the Tensor Data Mover (tensor_load_to_lds, TENSORcnt, double-
// buffered) and async loads (global_load_async_to_lds_b128, ASYNCcnt), with
// guarded fallbacks to the register-staged path.
#include <hip/hip_runtime.h>

typedef unsigned short ushort_t;
typedef __attribute__((ext_vector_type(16))) __bf16 v16bf;
typedef __attribute__((ext_vector_type(8)))  __bf16 v8bf;
typedef __attribute__((ext_vector_type(8)))  float  v8f;
typedef __attribute__((ext_vector_type(4)))  unsigned int u32x4;

#define DIM    1024
#define NH     16
#define HD     64
#define SEQ    2048
#define BATCH  2
#define HIDDEN 2816
#define TOKENS (BATCH*SEQ)   // 4096

// ---------- CDNA5 data-mover feature gates (device pass only) ----------
#if defined(__gfx1250__) && __has_builtin(__builtin_amdgcn_tensor_load_to_lds) && __has_builtin(__builtin_amdgcn_s_wait_tensorcnt)
#define HAS_TDM 1
#endif
#if defined(__gfx1250__) && __has_builtin(__builtin_amdgcn_global_load_async_to_lds_b128) && __has_builtin(__builtin_amdgcn_s_wait_asynccnt)
#define HAS_ASYNC 1
#endif

#if defined(HAS_ASYNC)
// Builtin expects int4 pointers; clang prints addrspace(1) as "__device__".
typedef __attribute__((ext_vector_type(4))) int i32x4v;
typedef __attribute__((address_space(1))) i32x4v as1_i4;
typedef __attribute__((address_space(3))) i32x4v as3_i4;
#define ASYNC_G(p) ((as1_i4*)(const void*)(p))
#define ASYNC_L(p) ((as3_i4*)(const void*)(p))
#endif

#if defined(HAS_TDM)
typedef __attribute__((ext_vector_type(4))) unsigned int tdm_v4u;
typedef __attribute__((ext_vector_type(8))) int tdm_v8i;
typedef __attribute__((ext_vector_type(4))) int tdm_v4i;

// Low 32 bits of a generic LDS pointer are the LDS byte offset (ISA 10.2).
__device__ __forceinline__ unsigned lds_u32(const void* p){
  return (unsigned)(unsigned long long)p;
}
// 2-D bf16 tile load via Tensor Data Mover. D# per ISA 8.3/8.4:
// group0: count=1, lds_addr, global tile address, type=2("image").
// group1: data_size=1 (2 bytes), tensor_dim0/1 = tile dims (tile fully in
//         bounds -> no clipping), tile_dim0/1, dim0 stride; dim2+ unused.
__device__ __forceinline__ void tdm_load_tile(unsigned lds_off, const ushort_t* gsrc,
    unsigned tile_w, unsigned tile_h, unsigned long long stride_elems)
{
  unsigned long long ga = (unsigned long long)gsrc;
  tdm_v4u g0;
  g0[0] = 1u;                                              // 1 valid descriptor
  g0[1] = lds_off;                                         // lds_addr (bytes)
  g0[2] = (unsigned)ga;                                    // global_addr[31:0]
  g0[3] = (unsigned)((ga >> 32) & 0x01FFFFFFull) | (2u << 30); // [56:32] | type=2
  tdm_v8i g1;
  g1[0] = (int)(1u << 16);                                 // data_size=1 (2B), wg_mask=0
  g1[1] = (int)((tile_w & 0xFFFFu) << 16);                 // tensor_dim0[15:0]
  g1[2] = (int)((tile_w >> 16) | ((tile_h & 0xFFFFu) << 16)); // t_d0[31:16]|t_d1[15:0]
  g1[3] = (int)((tile_h >> 16) | ((tile_w & 0xFFFFu) << 16)); // t_d1[31:16]|tile_dim0
  g1[4] = (int)(tile_h & 0xFFFFu);                         // tile_dim1; tile_dim2=0
  g1[5] = (int)(unsigned)(stride_elems & 0xFFFFFFFFull);   // dim0_stride[31:0]
  g1[6] = (int)(unsigned)((stride_elems >> 32) & 0xFFFFull); // [47:32]; dim1_stride=0
  g1[7] = 0;
  tdm_v4i gz; gz[0]=0; gz[1]=0; gz[2]=0; gz[3]=0;
#if __clang_major__ >= 23
  tdm_v8i gz8; gz8[0]=0; gz8[1]=0; gz8[2]=0; gz8[3]=0; gz8[4]=0; gz8[5]=0; gz8[6]=0; gz8[7]=0;
  __builtin_amdgcn_tensor_load_to_lds(g0, g1, gz, gz, gz8, 0);
#else
  __builtin_amdgcn_tensor_load_to_lds(g0, g1, gz, gz, 0);
#endif
}
#endif

// ---------------- bf16 helpers ----------------
__device__ __forceinline__ ushort_t f2bf(float f){
  unsigned u = __float_as_uint(f);
  u += 0x7FFFu + ((u >> 16) & 1u);          // round-to-nearest-even
  return (ushort_t)(u >> 16);
}
__device__ __forceinline__ float bf2f(ushort_t h){
  return __uint_as_float(((unsigned)h) << 16);
}
__device__ __forceinline__ v8f v8f_zero(){
  v8f z;
#pragma unroll
  for(int i=0;i<8;++i) z[i]=0.0f;
  return z;
}
// 16x16x32 bf16 fragment: lanes 0-15 hold K=k0..k0+7 / k0+16..k0+23,
// lanes 16-31 hold K=k0+8..k0+15 / k0+24..k0+31 (ISA 7.12.2).
__device__ __forceinline__ v16bf ldfrag(const ushort_t* rowptr, int k0){
  v8bf lo = *(const v8bf*)(rowptr + k0);
  v8bf hi = *(const v8bf*)(rowptr + k0 + 16);
  v16bf r;
#pragma unroll
  for(int i=0;i<8;++i){ r[i]=lo[i]; r[i+8]=hi[i]; }
  return r;
}
__device__ __forceinline__ v8f wmma_bf16(v16bf a, v16bf b, v8f c){
  return __builtin_amdgcn_wmma_f32_16x16x32_bf16(false, a, false, b, (short)0, c, false, false);
}

// ---------------- tiny prelim: t = silu(adafm_input) @ proj_w + proj_b ----------------
__global__ __launch_bounds__(256) void proj_t_kernel(const float* __restrict__ ain,
    const float* __restrict__ pw, const float* __restrict__ pb, float* __restrict__ t)
{
  int col = blockIdx.x*256 + threadIdx.x;     // 0..511
  int b   = blockIdx.y;
  float acc = 0.f;
  for(int k=0;k<DIM;++k){
    float a = ain[b*DIM + k];
    a = a / (1.0f + __expf(-a));
    acc += a * pw[k*512 + col];
  }
  t[b*512 + col] = acc + pb[col];
}

// circulant (transposed, N x K): circ[n*256+k] = g[(n-k) mod 256],
// g_d = (1/256) * sum_k f_k cos(2*pi*k*d/256)  (real part of IFFT of real f)
__global__ __launch_bounds__(256) void build_circ_kernel(const float* __restrict__ t,
    ushort_t* __restrict__ circ)
{
  int b = blockIdx.x, branch = blockIdx.y;
  __shared__ float fsh[256];
  __shared__ float g[256];
  int d = threadIdx.x;
  fsh[d] = t[b*512 + branch*256 + d];
  __syncthreads();
  float s = 0.f;
  const float w = 6.2831853071795864769f / 256.0f;
  for(int k=0;k<256;++k){
    int kd = (k*d) & 255;
    s += fsh[k] * __cosf((float)kd * w);
  }
  g[d] = s * (1.0f/256.0f);
  __syncthreads();
  ushort_t* outp = circ + ((size_t)branch*2 + b) * 65536;   // [branch][batch]
  for(int k=0;k<256;++k) outp[d*256 + k] = f2bf(g[(d - k) & 255]);
}

// ---------------- weight pack: fp32 (K x N) -> bf16 (N x K) ----------------
__global__ __launch_bounds__(256) void pack_wt_kernel(const float* __restrict__ W,
    ushort_t* __restrict__ Wt, int K, int N)
{
  __shared__ float tile[32][33];
  int nb = blockIdx.x*32, kb = blockIdx.y*32;
  int tx = threadIdx.x & 31, ty = threadIdx.x >> 5;   // ty 0..7
#pragma unroll
  for(int i=0;i<32;i+=8) tile[ty+i][tx] = W[(size_t)(kb+ty+i)*N + nb + tx];
  __syncthreads();
#pragma unroll
  for(int i=0;i<32;i+=8) Wt[(size_t)(nb+ty+i)*K + kb + tx] = f2bf(tile[tx][ty+i]);
}

// ---------------- generic bf16 WMMA GEMM: C(MxN) = A(MxK,bf16) * Bt(NxK,bf16)^T ----------------
template<bool OUT_BF16>
__global__ __launch_bounds__(256) void gemm_bf16_kernel(const ushort_t* __restrict__ A,
    const ushort_t* __restrict__ Bt, void* __restrict__ Cv, int M, int N, int K)
{
  constexpr int BM=128, BN=128, BK=32;
#if defined(HAS_TDM)
  __shared__ __align__(16) ushort_t As[2][BM*BK];   // double-buffered TDM tiles
  __shared__ __align__(16) ushort_t Bs[2][BN*BK];
#else
  __shared__ __align__(16) ushort_t As[1][BM*BK];
  __shared__ __align__(16) ushort_t Bs[1][BN*BK];
#endif
  const int tid  = threadIdx.x;
  const int wave = tid >> 5, lane = tid & 31;
  const int lm = lane & 15, hb8 = (lane >> 4) << 3, rbase = (lane >> 4) * 8;
  const int wm = (wave & 3) * 32, wn = (wave >> 2) * 64;   // 4x2 wave grid
  const int m0 = blockIdx.y * BM, n0 = blockIdx.x * BN;

  v8f acc[2][4];
#pragma unroll
  for(int i=0;i<2;++i)
#pragma unroll
    for(int j=0;j<4;++j) acc[i][j] = v8f_zero();

#if defined(HAS_TDM)
  // Tensor Data Mover pipeline: wave0 DMAs tiles, double-buffered.
  const int nIter = K / BK;
  if(wave == 0){
    tdm_load_tile(lds_u32(As[0]), A  + (size_t)m0*K, BK, BM, (unsigned long long)K);
    tdm_load_tile(lds_u32(Bs[0]), Bt + (size_t)n0*K, BK, BN, (unsigned long long)K);
  }
  for(int it=0; it<nIter; ++it){
    const int k0 = it*BK;
    if(wave == 0){
      if(it+1 < nIter){
        const int nb = (it+1) & 1;
        tdm_load_tile(lds_u32(As[nb]), A  + (size_t)m0*K + (k0+BK), BK, BM, (unsigned long long)K);
        tdm_load_tile(lds_u32(Bs[nb]), Bt + (size_t)n0*K + (k0+BK), BK, BN, (unsigned long long)K);
        __builtin_amdgcn_s_wait_tensorcnt(2);   // current pair done, next pair in flight
      } else {
        __builtin_amdgcn_s_wait_tensorcnt(0);
      }
    }
    __syncthreads();
    const ushort_t* Ac = As[it & 1];
    const ushort_t* Bc = Bs[it & 1];
    v16bf af[2], bfr[4];
#pragma unroll
    for(int mi=0;mi<2;++mi) af[mi]  = ldfrag(Ac + (wm + mi*16 + lm)*BK, hb8);
#pragma unroll
    for(int ni=0;ni<4;++ni) bfr[ni] = ldfrag(Bc + (wn + ni*16 + lm)*BK, hb8);
#pragma unroll
    for(int mi=0;mi<2;++mi)
#pragma unroll
      for(int ni=0;ni<4;++ni)
        acc[mi][ni] = wmma_bf16(af[mi], bfr[ni], acc[mi][ni]);
    __syncthreads();
  }
#else
  const int arow = tid >> 1, aseg = (tid & 1) * 16;
  for(int k0=0;k0<K;k0+=BK){
#if defined(HAS_ASYNC)
    __builtin_amdgcn_global_load_async_to_lds_b128(
        ASYNC_G(A  + (size_t)(m0+arow)*K + k0 + aseg),     ASYNC_L(As[0] + arow*BK + aseg),     0, 0);
    __builtin_amdgcn_global_load_async_to_lds_b128(
        ASYNC_G(A  + (size_t)(m0+arow)*K + k0 + aseg + 8), ASYNC_L(As[0] + arow*BK + aseg + 8), 0, 0);
    __builtin_amdgcn_global_load_async_to_lds_b128(
        ASYNC_G(Bt + (size_t)(n0+arow)*K + k0 + aseg),     ASYNC_L(Bs[0] + arow*BK + aseg),     0, 0);
    __builtin_amdgcn_global_load_async_to_lds_b128(
        ASYNC_G(Bt + (size_t)(n0+arow)*K + k0 + aseg + 8), ASYNC_L(Bs[0] + arow*BK + aseg + 8), 0, 0);
    __builtin_amdgcn_s_wait_asynccnt(0);
#else
    const u32x4* sA = (const u32x4*)(A  + (size_t)(m0+arow)*K + k0 + aseg);
    const u32x4* sB = (const u32x4*)(Bt + (size_t)(n0+arow)*K + k0 + aseg);
    u32x4 a0=sA[0], a1=sA[1], b0=sB[0], b1=sB[1];
    if(k0 + BK < K){
      __builtin_prefetch(A  + (size_t)(m0+arow)*K + k0 + BK + aseg, 0, 0);
      __builtin_prefetch(Bt + (size_t)(n0+arow)*K + k0 + BK + aseg, 0, 0);
    }
    u32x4* dA = (u32x4*)(As[0] + arow*BK + aseg); dA[0]=a0; dA[1]=a1;
    u32x4* dB = (u32x4*)(Bs[0] + arow*BK + aseg); dB[0]=b0; dB[1]=b1;
#endif
    __syncthreads();
    v16bf af[2], bfr[4];
#pragma unroll
    for(int mi=0;mi<2;++mi) af[mi]  = ldfrag(As[0] + (wm + mi*16 + lm)*BK, hb8);
#pragma unroll
    for(int ni=0;ni<4;++ni) bfr[ni] = ldfrag(Bs[0] + (wn + ni*16 + lm)*BK, hb8);
#pragma unroll
    for(int mi=0;mi<2;++mi)
#pragma unroll
      for(int ni=0;ni<4;++ni)
        acc[mi][ni] = wmma_bf16(af[mi], bfr[ni], acc[mi][ni]);
    __syncthreads();
  }
#endif
#pragma unroll
  for(int mi=0;mi<2;++mi)
#pragma unroll
    for(int ni=0;ni<4;++ni)
#pragma unroll
      for(int r=0;r<8;++r){
        int row = m0 + wm + mi*16 + rbase + r;
        int col = n0 + wn + ni*16 + lm;
        float v = acc[mi][ni][r];
        if(OUT_BF16) ((ushort_t*)Cv)[(size_t)row*N + col] = f2bf(v);
        else         ((float*)Cv)[(size_t)row*N + col]    = v;
      }
}

// ---------------- AdaFM modulate: per-patch circular conv as WMMA GEMM ----------------
__global__ __launch_bounds__(256) void modulate_kernel(const float* __restrict__ X,
    const ushort_t* __restrict__ circ, ushort_t* __restrict__ Y)
{
  constexpr int BM=128, BN=128, BK=32, KK=256;
  __shared__ __align__(16) ushort_t As[BM*BK];
  __shared__ __align__(16) ushort_t Bs[BN*BK];
  const int batch = blockIdx.z;
  const float*   xb = X + (size_t)batch*SEQ*DIM;
  ushort_t*      yb = Y + (size_t)batch*SEQ*DIM;
  const ushort_t* Bt = circ + (size_t)batch*KK*KK;

  const int tid = threadIdx.x;
  const int wave = tid >> 5, lane = tid & 31;
  const int lm = lane & 15, hb8 = (lane >> 4) << 3, rbase = (lane >> 4) * 8;
  const int wm = (wave & 3) * 32, wn = (wave >> 2) * 64;
  const int m0 = blockIdx.y * BM, n0 = blockIdx.x * BN;
  const int pl = tid >> 1, seg = (tid & 1) * 16;

  v8f acc[2][4];
#pragma unroll
  for(int i=0;i<2;++i)
#pragma unroll
    for(int j=0;j<4;++j) acc[i][j] = v8f_zero();

  const int p_load = m0 + pl;               // patch gathered by this thread
  const int brL = p_load >> 6, bcL = p_load & 63;

  for(int k0=0;k0<KK;k0+=BK){
    int mbase = k0 + seg;                   // element index inside 16x16 patch
    const float4* s = (const float4*)(xb + (size_t)(brL*16 + (mbase>>4))*DIM + bcL*16);
    ushort_t* dstp = As + pl*BK + seg;
#pragma unroll
    for(int i=0;i<4;++i){
      float4 v = s[i];
      dstp[4*i+0]=f2bf(v.x); dstp[4*i+1]=f2bf(v.y);
      dstp[4*i+2]=f2bf(v.z); dstp[4*i+3]=f2bf(v.w);
    }
    const u32x4* sB = (const u32x4*)(Bt + (size_t)(n0+pl)*KK + k0 + seg);
    u32x4* dB = (u32x4*)(Bs + pl*BK + seg);
    dB[0]=sB[0]; dB[1]=sB[1];
    __syncthreads();
    v16bf af[2], bfr[4];
#pragma unroll
    for(int mi=0;mi<2;++mi) af[mi]  = ldfrag(As + (wm + mi*16 + lm)*BK, hb8);
#pragma unroll
    for(int ni=0;ni<4;++ni) bfr[ni] = ldfrag(Bs + (wn + ni*16 + lm)*BK, hb8);
#pragma unroll
    for(int mi=0;mi<2;++mi)
#pragma unroll
      for(int ni=0;ni<4;++ni)
        acc[mi][ni] = wmma_bf16(af[mi], bfr[ni], acc[mi][ni]);
    __syncthreads();
  }
#pragma unroll
  for(int mi=0;mi<2;++mi)
#pragma unroll
    for(int ni=0;ni<4;++ni)
#pragma unroll
      for(int r=0;r<8;++r){
        int p  = m0 + wm + mi*16 + rbase + r;
        int br = p >> 6, bc = p & 63;
        int n  = n0 + wn + ni*16 + lm;
        yb[(size_t)(br*16 + (n>>4))*DIM + bc*16 + (n&15)] = f2bf(acc[mi][ni][r]);
      }
}

// ---------------- rotary + per-head justnorm*sqk (in place) + V transpose ----------------
__global__ __launch_bounds__(256) void qkv_post_kernel(ushort_t* __restrict__ QKV,
    ushort_t* __restrict__ Vt, const float* __restrict__ fcos, const float* __restrict__ fsin,
    const float* __restrict__ sqk)
{
  int g = threadIdx.x >> 5, lane = threadIdx.x & 31;
  int idx   = blockIdx.x * 8 + g;           // (token, head)
  int h     = idx & 15;
  int token = idx >> 4;
  int s     = token & (SEQ-1);
  int b     = token >> 11;
  size_t rowoff = (size_t)token*DIM + h*HD;
  float c  = fcos[s*32 + lane], sn = fsin[s*32 + lane];
  float se0 = sqk[h*HD + 2*lane]   * 32.0f; // sqrt(DIM)=32
  float se1 = sqk[h*HD + 2*lane+1] * 32.0f;
#pragma unroll
  for(int which=0;which<2;++which){         // 0 = Q, 1 = K
    ushort_t* p = QKV + (size_t)which*TOKENS*DIM + rowoff;
    float a  = bf2f(p[2*lane]);
    float bb = bf2f(p[2*lane+1]);
    float ra = a*c - bb*sn;
    float rb = a*sn + bb*c;
    float ss = ra*ra + rb*rb;
#pragma unroll
    for(int d=16; d>=1; d>>=1) ss += __shfl_xor(ss, d, 32);
    float inv = 1.0f / fmaxf(sqrtf(ss), 1e-12f);
    p[2*lane]   = f2bf(ra*inv*se0);
    p[2*lane+1] = f2bf(rb*inv*se1);
  }
  const ushort_t* vp = QKV + (size_t)2*TOKENS*DIM + rowoff;
  size_t vtb = ((size_t)(b*NH + h)*HD)*SEQ;
  Vt[vtb + (size_t)(2*lane)*SEQ + s]   = vp[2*lane];
  Vt[vtb + (size_t)(2*lane+1)*SEQ + s] = vp[2*lane+1];
}

// ---------------- flash attention (full, no mask): per (b, h, 128 q-rows) ----------------
__global__ __launch_bounds__(256) void attention_kernel(const ushort_t* __restrict__ Q,
    const ushort_t* __restrict__ Kc, const ushort_t* __restrict__ Vt,
    ushort_t* __restrict__ Ctx)
{
  __shared__ __align__(16) ushort_t Ks[32*64];     // 32 keys x 64 hd
  __shared__ __align__(16) ushort_t Vs[64*32];     // 64 hd x 32 keys (pre-transposed V)
  __shared__ __align__(16) ushort_t Ps[8*16*32];   // per-wave P scratch
  int bid = blockIdx.x;
  int qb = bid & 15, h = (bid >> 4) & 15, b = bid >> 8;
  int tid = threadIdx.x, wave = tid >> 5, lane = tid & 31;
  int lm = lane & 15, hb8 = (lane >> 4) << 3, rbase = (lane >> 4) * 8;

  int qrow = qb*128 + wave*16 + lm;
  const ushort_t* qp = Q + (size_t)(b*SEQ + qrow)*DIM + h*HD;
  v16bf aq0 = ldfrag(qp, hb8);          // head-dim 0..31
  v16bf aq1 = ldfrag(qp + 32, hb8);     // head-dim 32..63

  v8f o[4];
  float mI[8], lI[8];
#pragma unroll
  for(int ni=0;ni<4;++ni) o[ni] = v8f_zero();
#pragma unroll
  for(int r=0;r<8;++r){ mI[r] = -3.0e38f; lI[r] = 0.0f; }

  const ushort_t* kbase = Kc + (size_t)(b*SEQ)*DIM + h*HD;
  const ushort_t* vbase = Vt + ((size_t)(b*NH + h)*HD)*SEQ;

  for(int kb=0; kb<SEQ; kb+=32){
#if defined(HAS_ASYNC)
    { int r = tid >> 3, seg = (tid & 7) * 8;
      __builtin_amdgcn_global_load_async_to_lds_b128(
          ASYNC_G(kbase + (size_t)(kb + r)*DIM + seg), ASYNC_L(Ks + r*64 + seg), 0, 0); }
    { int r = tid >> 2, seg = (tid & 3) * 8;
      __builtin_amdgcn_global_load_async_to_lds_b128(
          ASYNC_G(vbase + (size_t)r*SEQ + kb + seg),   ASYNC_L(Vs + r*32 + seg), 0, 0); }
    __builtin_amdgcn_s_wait_asynccnt(0);
#elif defined(HAS_TDM)
    if(wave == 0){
      tdm_load_tile(lds_u32(Ks), kbase + (size_t)kb*DIM, 64, 32, (unsigned long long)DIM);
      tdm_load_tile(lds_u32(Vs), vbase + kb,             32, 64, (unsigned long long)SEQ);
      __builtin_amdgcn_s_wait_tensorcnt(0);
    }
#else
    { int r = tid >> 3, seg = (tid & 7) * 8;
      *(u32x4*)(Ks + r*64 + seg) = *(const u32x4*)(kbase + (size_t)(kb + r)*DIM + seg); }
    { int r = tid >> 2, seg = (tid & 3) * 8;
      *(u32x4*)(Vs + r*32 + seg) = *(const u32x4*)(vbase + (size_t)r*SEQ + kb + seg); }
#endif
    __syncthreads();

    v8f s0 = v8f_zero(), s1 = v8f_zero();
    v16bf bk;
    bk = ldfrag(Ks + lm*64,            hb8); s0 = wmma_bf16(aq0, bk, s0);
    bk = ldfrag(Ks + lm*64 + 32,       hb8); s0 = wmma_bf16(aq1, bk, s0);
    bk = ldfrag(Ks + (16+lm)*64,       hb8); s1 = wmma_bf16(aq0, bk, s1);
    bk = ldfrag(Ks + (16+lm)*64 + 32,  hb8); s1 = wmma_bf16(aq1, bk, s1);

    ushort_t* ps = Ps + wave*512;
#pragma unroll
    for(int r=0;r<8;++r){
      float x0 = s0[r]*8.0f, x1 = s1[r]*8.0f;     // scale = sqrt(HD) = 8
      float mx = fmaxf(x0, x1);
#pragma unroll
      for(int d=8; d>=1; d>>=1) mx = fmaxf(mx, __shfl_xor(mx, d, 16));
      float mn   = fmaxf(mI[r], mx);
      float corr = __expf(mI[r] - mn);
      float p0 = __expf(x0 - mn), p1 = __expf(x1 - mn);
      float rs = p0 + p1;
#pragma unroll
      for(int d=8; d>=1; d>>=1) rs += __shfl_xor(rs, d, 16);
      lI[r] = lI[r]*corr + rs;
      mI[r] = mn;
#pragma unroll
      for(int ni=0;ni<4;++ni) o[ni][r] *= corr;
      int prow = rbase + r;
      ps[prow*32 + lm]      = f2bf(p0);
      ps[prow*32 + 16 + lm] = f2bf(p1);
    }
    __builtin_amdgcn_wave_barrier();    // wave-private LDS round trip; DS ops in-order
    v16bf ap = ldfrag(ps + lm*32, hb8);
#pragma unroll
    for(int ni=0;ni<4;++ni){
      v16bf bv = ldfrag(Vs + (ni*16 + lm)*32, hb8);
      o[ni] = wmma_bf16(ap, bv, o[ni]);
    }
    __syncthreads();
  }
#pragma unroll
  for(int ni=0;ni<4;++ni)
#pragma unroll
    for(int r=0;r<8;++r){
      int row = qb*128 + wave*16 + rbase + r;
      int col = h*HD + ni*16 + lm;
      float inv = 1.0f / lI[r];
      Ctx[(size_t)(b*SEQ + row)*DIM + col] = f2bf(o[ni][r]*inv);
    }
}

// ---------------- residual + renorm: out = justnorm(h + |alpha*1.6|*(justnorm(hb)-h)) ----------------
__device__ __forceinline__ float block_sum(float v, float* red){
#pragma unroll
  for(int d=16; d>=1; d>>=1) v += __shfl_xor(v, d, 32);
  int w = threadIdx.x >> 5;
  __syncthreads();
  if((threadIdx.x & 31) == 0) red[w] = v;
  __syncthreads();
  float r = 0.f;
#pragma unroll
  for(int i=0;i<8;++i) r += red[i];
  return r;
}

__global__ __launch_bounds__(256) void resid_norm_kernel(const float* __restrict__ Xin,
    const float* __restrict__ Hb, const float* __restrict__ alpha, float* __restrict__ Out)
{
  __shared__ float red[8];
  int row = blockIdx.x;
  const float* xr = Xin + (size_t)row*DIM;
  const float* hr = Hb  + (size_t)row*DIM;
  float x[4], hb[4];
  float s1 = 0.f, s2 = 0.f;
#pragma unroll
  for(int j=0;j<4;++j){
    int d = threadIdx.x + j*256;
    x[j] = xr[d]; hb[j] = hr[d];
    s1 += x[j]*x[j]; s2 += hb[j]*hb[j];
  }
  s1 = block_sum(s1, red);
  s2 = block_sum(s2, red);
  float i1 = 1.0f / fmaxf(sqrtf(s1), 1e-12f);
  float i2 = 1.0f / fmaxf(sqrtf(s2), 1e-12f);
  float rr[4]; float s3 = 0.f;
#pragma unroll
  for(int j=0;j<4;++j){
    int d = threadIdx.x + j*256;
    float h  = x[j]*i1;
    float hn = hb[j]*i2;
    float lr = fabsf(alpha[d]*1.6f);      // 0.05*sqrt(DIM)
    rr[j] = h + lr*(hn - h);
    s3 += rr[j]*rr[j];
  }
  s3 = block_sum(s3, red);
  float i3 = 1.0f / fmaxf(sqrtf(s3), 1e-12f);
#pragma unroll
  for(int j=0;j<4;++j){
    int d = threadIdx.x + j*256;
    Out[(size_t)row*DIM + d] = rr[j]*i3;
  }
}

// ---------------- FFN activation: silu(h1*sv*sqrt(HIDDEN)) * (h3*su) ----------------
__global__ __launch_bounds__(256) void ffn_act_kernel(const ushort_t* __restrict__ h1,
    const ushort_t* __restrict__ h3, const float* __restrict__ su,
    const float* __restrict__ sv, ushort_t* __restrict__ outp)
{
  size_t idx = (size_t)blockIdx.x*256 + threadIdx.x;
  int col = (int)(idx % HIDDEN);
  float z = bf2f(h1[idx]) * sv[col] * 53.0660171f;   // sqrt(2816)
  float u = bf2f(h3[idx]) * su[col];
  float sil = z / (1.0f + __expf(-z));
  outp[idx] = f2bf(sil * u);
}

// ============================ host orchestration ============================
extern "C" void kernel_launch(void* const* d_in, const int* in_sizes, int n_in,
                              void* d_out, int out_size, void* d_ws, size_t ws_size,
                              hipStream_t stream)
{
  (void)in_sizes; (void)n_in; (void)out_size; (void)ws_size;
  const float* x      = (const float*)d_in[0];
  const float* fcos   = (const float*)d_in[1];
  const float* fsin   = (const float*)d_in[2];
  const float* a_in   = (const float*)d_in[3];
  const float* wq     = (const float*)d_in[4];
  const float* wk     = (const float*)d_in[5];
  const float* wv     = (const float*)d_in[6];
  const float* wo     = (const float*)d_in[7];
  const float* sqk    = (const float*)d_in[8];
  const float* w1     = (const float*)d_in[9];
  const float* w2     = (const float*)d_in[10];
  const float* w3     = (const float*)d_in[11];
  const float* su     = (const float*)d_in[12];
  const float* sv     = (const float*)d_in[13];
  const float* projw  = (const float*)d_in[14];
  const float* projb  = (const float*)d_in[15];
  const float* aalpha = (const float*)d_in[16];
  const float* malpha = (const float*)d_in[17];
  float* out = (float*)d_out;

  char* ws = (char*)d_ws;
  size_t off = 0;
  auto alloc = [&](size_t bytes)->char*{
    char* p = ws + off; off += (bytes + 255) & ~(size_t)255; return p;
  };
  float*    t_buf = (float*)   alloc((size_t)2*512*sizeof(float));
  ushort_t* circ  = (ushort_t*)alloc((size_t)4*256*256*2);        // [branch][batch]
  ushort_t* wq_t  = (ushort_t*)alloc((size_t)DIM*DIM*2);
  ushort_t* wk_t  = (ushort_t*)alloc((size_t)DIM*DIM*2);
  ushort_t* wv_t  = (ushort_t*)alloc((size_t)DIM*DIM*2);
  ushort_t* wo_t  = (ushort_t*)alloc((size_t)DIM*DIM*2);
  ushort_t* w1_t  = (ushort_t*)alloc((size_t)HIDDEN*DIM*2);
  ushort_t* w3_t  = (ushort_t*)alloc((size_t)HIDDEN*DIM*2);
  ushort_t* w2_t  = (ushort_t*)alloc((size_t)DIM*HIDDEN*2);
  ushort_t* xm    = (ushort_t*)alloc((size_t)TOKENS*DIM*2);
  ushort_t* qkv   = (ushort_t*)alloc((size_t)3*TOKENS*DIM*2);
  ushort_t* vt    = (ushort_t*)alloc((size_t)TOKENS*DIM*2);
  ushort_t* ctx   = (ushort_t*)alloc((size_t)TOKENS*DIM*2);
  float*    hbuf  = (float*)   alloc((size_t)TOKENS*DIM*4);       // h_a then h_m
  float*    x1    = (float*)   alloc((size_t)TOKENS*DIM*4);
  ushort_t* h1    = (ushort_t*)alloc((size_t)TOKENS*HIDDEN*2);
  ushort_t* h3    = (ushort_t*)alloc((size_t)TOKENS*HIDDEN*2);
  ushort_t* hact  = (ushort_t*)alloc((size_t)TOKENS*HIDDEN*2);

  dim3 B256(256);
  // prelims
  proj_t_kernel<<<dim3(2,2), B256, 0, stream>>>(a_in, projw, projb, t_buf);
  build_circ_kernel<<<dim3(2,2), B256, 0, stream>>>(t_buf, circ);
  // weight packs (bf16, N x K)
  pack_wt_kernel<<<dim3(DIM/32,    DIM/32),    B256, 0, stream>>>(wq, wq_t, DIM, DIM);
  pack_wt_kernel<<<dim3(DIM/32,    DIM/32),    B256, 0, stream>>>(wk, wk_t, DIM, DIM);
  pack_wt_kernel<<<dim3(DIM/32,    DIM/32),    B256, 0, stream>>>(wv, wv_t, DIM, DIM);
  pack_wt_kernel<<<dim3(DIM/32,    DIM/32),    B256, 0, stream>>>(wo, wo_t, DIM, DIM);
  pack_wt_kernel<<<dim3(HIDDEN/32, DIM/32),    B256, 0, stream>>>(w1, w1_t, DIM, HIDDEN);
  pack_wt_kernel<<<dim3(HIDDEN/32, DIM/32),    B256, 0, stream>>>(w3, w3_t, DIM, HIDDEN);
  pack_wt_kernel<<<dim3(DIM/32,    HIDDEN/32), B256, 0, stream>>>(w2, w2_t, HIDDEN, DIM);
  // attention branch
  modulate_kernel<<<dim3(2,64,2), B256, 0, stream>>>(x, circ, xm);
  gemm_bf16_kernel<true ><<<dim3(DIM/128, TOKENS/128), B256, 0, stream>>>(xm, wq_t, qkv,                        TOKENS, DIM, DIM);
  gemm_bf16_kernel<true ><<<dim3(DIM/128, TOKENS/128), B256, 0, stream>>>(xm, wk_t, qkv + (size_t)TOKENS*DIM,   TOKENS, DIM, DIM);
  gemm_bf16_kernel<true ><<<dim3(DIM/128, TOKENS/128), B256, 0, stream>>>(xm, wv_t, qkv + (size_t)2*TOKENS*DIM, TOKENS, DIM, DIM);
  qkv_post_kernel<<<dim3(8192), B256, 0, stream>>>(qkv, vt, fcos, fsin, sqk);
  attention_kernel<<<dim3(512), B256, 0, stream>>>(qkv, qkv + (size_t)TOKENS*DIM, vt, ctx);
  gemm_bf16_kernel<false><<<dim3(DIM/128, TOKENS/128), B256, 0, stream>>>(ctx, wo_t, hbuf, TOKENS, DIM, DIM);
  resid_norm_kernel<<<dim3(TOKENS), B256, 0, stream>>>(x, hbuf, aalpha, x1);
  // MLP branch
  modulate_kernel<<<dim3(2,64,2), B256, 0, stream>>>(x1, circ + (size_t)2*65536, xm);
  gemm_bf16_kernel<true ><<<dim3(HIDDEN/128, TOKENS/128), B256, 0, stream>>>(xm, w1_t, h1, TOKENS, HIDDEN, DIM);
  gemm_bf16_kernel<true ><<<dim3(HIDDEN/128, TOKENS/128), B256, 0, stream>>>(xm, w3_t, h3, TOKENS, HIDDEN, DIM);
  ffn_act_kernel<<<dim3((TOKENS*HIDDEN)/256), B256, 0, stream>>>(h1, h3, su, sv, hact);
  gemm_bf16_kernel<false><<<dim3(DIM/128, TOKENS/128), B256, 0, stream>>>(hact, w2_t, hbuf, TOKENS, DIM, HIDDEN);
  resid_norm_kernel<<<dim3(TOKENS), B256, 0, stream>>>(x1, hbuf, malpha, out);
}
